// AdaptiveConv2d_31507880083618
// MI455X (gfx1250) — matrix operand order repaired and checked by reading, not compiled, not run on hardware
//
#include <hip/hip_runtime.h>

typedef __bf16 bf16_t;
typedef __attribute__((ext_vector_type(16))) __bf16 v16bf;
typedef __attribute__((ext_vector_type(8)))  float  v8f;
typedef __attribute__((ext_vector_type(4)))  unsigned int v4u;
typedef int v4i_vs __attribute__((vector_size(16)));   // matches builtin param pointee

#define B_    16
#define CIN   64
#define COUT  64
#define HH    128
#define WW    128
#define NKW   (COUT * CIN * 3 * 3)   /* 36864 */
#define ROWS  (NKW + COUT)           /* 36928 */

/* workspace layout (bytes):
   [0, 1179648)            : pre-swizzled bf16 A-fragments (1152 frags * 1024 B)
   [1179648, 1183744)      : per-batch biases, 16*64 f32
   [1183744, 1185664)      : h1 activations, 16*30 f32
   [1185792, +33554432)    : xt = x transposed to [b][h][w][ci], bf16           */
#define WS_BIAS_OFF 1179648
#define WS_H1_OFF   1183744
#define WS_XT_OFF   1185792

__device__ __forceinline__ bf16_t to_bf16(float f) {
    unsigned u = __builtin_bit_cast(unsigned, f);
    u += 0x7FFFu + ((u >> 16) & 1u);             // round-to-nearest-even
    unsigned short hi = (unsigned short)(u >> 16);
    return __builtin_bit_cast(bf16_t, hi);
}

/* ---- CDNA5 async global->LDS copy (ASYNCcnt-tracked), builtin or asm ---- */
__device__ __forceinline__ void async_copy_b128(const bf16_t* gsrc, bf16_t* ldst) {
#if __has_builtin(__builtin_amdgcn_global_load_async_to_lds_b128)
    __builtin_amdgcn_global_load_async_to_lds_b128(
        (__attribute__((address_space(1))) v4i_vs*)(void*)gsrc,
        (__attribute__((address_space(3))) v4i_vs*)(void*)ldst, 0, 0);
#else
    unsigned laddr = (unsigned)(size_t)(__attribute__((address_space(3))) bf16_t*)ldst;
    unsigned long long gaddr = (unsigned long long)(size_t)gsrc;
    asm volatile("global_load_async_to_lds_b128 %0, %1, off"
                 :: "v"(laddr), "v"(gaddr) : "memory");
#endif
}

__device__ __forceinline__ void wait_async0() {
#if __has_builtin(__builtin_amdgcn_s_wait_asynccnt)
    __builtin_amdgcn_s_wait_asynccnt(0);
#else
    asm volatile("s_wait_asynccnt 0" ::: "memory");
#endif
}

/* ---- stage 0: transpose+convert x to bf16 [b][h][w][ci] (one pass) ---- */
__global__ void x_to_bf16_t(const float* __restrict__ x, bf16_t* __restrict__ xt) {
    __shared__ __align__(16) bf16_t ls[WW][72];
    const int blk = blockIdx.x;
    const int b = blk >> 7, h = blk & 127;
    const int tid = threadIdx.x;

    const int ci = tid >> 2;         // 0..63
    const int q  = tid & 3;          // 32-wide w chunk
    const float4* xp = (const float4*)(x + (((long)(b * CIN + ci) * HH + h) * WW) + q * 32);
    #pragma unroll
    for (int c = 0; c < 8; ++c) {
        float4 f = xp[c];
        int w0 = q * 32 + c * 4;
        ls[w0 + 0][ci] = to_bf16(f.x);
        ls[w0 + 1][ci] = to_bf16(f.y);
        ls[w0 + 2][ci] = to_bf16(f.z);
        ls[w0 + 3][ci] = to_bf16(f.w);
    }
    __syncthreads();

    const int w = tid >> 1, half = tid & 1;
    const v4u* src = (const v4u*)&ls[w][half * 32];
    v4u* dst = (v4u*)(xt + (((long)(b * HH + h) * WW + w) * CIN) + half * 32);
    #pragma unroll
    for (int c = 0; c < 4; ++c) dst[c] = src[c];
}

/* ---- stage 1: tiny MLP (z -> h0 -> h1), one block ---- */
__global__ void hyper_mlp(const float* __restrict__ z,
                          const float* __restrict__ W0, const float* __restrict__ b0,
                          const float* __restrict__ W1, const float* __restrict__ b1,
                          float* __restrict__ h1out) {
    __shared__ float h0s[16 * 20];
    const int t = threadIdx.x;
    if (t < 16 * 20) {
        int b = t / 20, j = t % 20;
        float s = b0[j];
        #pragma unroll
        for (int k = 0; k < 16; ++k) s += z[b * 16 + k] * W0[j * 16 + k];
        h0s[t] = s > 0.f ? s : 0.f;
    }
    __syncthreads();
    if (t < 16 * 30) {
        int b = t / 30, j = t % 30;
        float s = b1[j];
        #pragma unroll
        for (int k = 0; k < 20; ++k) s += h0s[b * 20 + k] * W1[j * 20 + k];
        h1out[b * 30 + j] = s > 0.f ? s : 0.f;
    }
}

/* ---- stage 2: hypernet head; scatter weights into WMMA A-fragment layout ---- */
__global__ void hyper_head(const float* __restrict__ W2, const float* __restrict__ b2,
                           const float* __restrict__ h1, bf16_t* __restrict__ wsw,
                           float* __restrict__ biasws) {
    const int idx = blockIdx.x * 256 + threadIdx.x;      // exactly B_*ROWS threads
    const int b = idx / ROWS;
    const int r = idx - b * ROWS;
    const float* hb = h1 + b * 30;
    const float* wr = W2 + (long)r * 30;
    float s = b2[r];
    #pragma unroll
    for (int k = 0; k < 30; ++k) s += hb[k] * wr[k];
    s = s > 0.f ? s : 0.f;

    if (r < NKW) {
        int co  = r / 576;
        int rem = r - co * 576;
        int ci  = rem / 9;
        int p   = rem - ci * 9;
        int kh  = p / 3, kw = p - kh * 3;
        int mt = co >> 4, m  = co & 15;
        int kb = ci >> 5, cc = ci & 31;
        int hh  = cc >> 4;
        int run = (cc >> 3) & 1;
        int pos = cc & 7;
        int lane = m + 16 * hh;
        int e    = run * 8 + pos;
        int frag = (((b * 9 + kh * 3 + kw) * 4 + mt) * 2 + kb);
        wsw[(long)frag * 512 + lane * 16 + e] = to_bf16(s);
    } else {
        biasws[b * COUT + (r - NKW)] = s;
    }
}

/* ---- stage 3: per-sample conv as 9 shifted bf16 WMMA GEMMs ----
   grid = B*H workgroups; block = 256 threads = 8 wave32.
   Tile fill = contiguous async b128 copies from the pre-transposed xt tensor.  */
__launch_bounds__(256)
__global__ void adaptive_conv_wmma(const bf16_t* __restrict__ xt,
                                   const bf16_t* __restrict__ wsw,
                                   const float* __restrict__ biasws,
                                   float* __restrict__ out) {
    __shared__ __align__(16) bf16_t xs[3][132][72];   // [row][wp][ci], wp = w+1

    const int blk = blockIdx.x;
    const int b   = blk >> 7;        // /128
    const int h   = blk & 127;
    const int tid = threadIdx.x;

    /* zero halo columns wp=0 and wp=129 for all 3 rows: 6 slabs * 36 dwords */
    if (tid < 216) {
        int slab = tid / 36, j = tid - slab * 36;
        int r = slab >> 1;
        int wp = (slab & 1) ? 129 : 0;
        ((unsigned int*)&xs[r][wp][0])[j] = 0u;
    }
    /* out-of-image boundary rows fully zeroed (uniform branch per block) */
    if (h == 0)
        for (int i = tid; i < (132 * 72) / 2; i += 256)
            ((unsigned int*)&xs[0][0][0])[i] = 0u;
    if (h == HH - 1)
        for (int i = tid; i < (132 * 72) / 2; i += 256)
            ((unsigned int*)&xs[2][0][0])[i] = 0u;

    /* async-fill interior: per row 512 b128 chunks (w, ci-quarter) */
    for (int r = 0; r < 3; ++r) {
        int hr = h + r - 1;
        if (hr >= 0 && hr < HH) {
            const bf16_t* rowbase = xt + ((long)(b * HH + hr) * WW) * CIN;
            #pragma unroll
            for (int jj = 0; jj < 2; ++jj) {
                int j   = tid * 2 + jj;          // 0..511
                int w   = j >> 2;
                int ci0 = (j & 3) * 16;
                async_copy_b128(rowbase + w * CIN + ci0, &xs[r][w + 1][ci0]);
            }
        }
    }
    wait_async0();
    __syncthreads();

    const int lane = tid & 31;
    const int wv   = tid >> 5;       // 0..7
    const int mt   = wv & 3;         // output-channel tile 0..3
    const int ntg  = wv >> 2;        // 0..1 : which group of 4 N-tiles
    const int n    = lane & 15;
    const int hh   = lane >> 4;

    v8f acc[4] = {v8f{}, v8f{}, v8f{}, v8f{}};

    #pragma unroll
    for (int kh = 0; kh < 3; ++kh) {
        #pragma unroll
        for (int kw = 0; kw < 3; ++kw) {
            #pragma unroll
            for (int kb = 0; kb < 2; ++kb) {
                /* A fragment: one lane = 32 contiguous bytes, pre-swizzled in ws */
                const int frag = (((b * 9 + kh * 3 + kw) * 4 + mt) * 2 + kb);
                union { v4u u[2]; v16bf v; } au;
                const v4u* pa = (const v4u*)(wsw + (long)frag * 512 + lane * 16);
                au.u[0] = pa[0];
                au.u[1] = pa[1];

                const int cib = kb * 32 + hh * 16;   // 16 consecutive ci for B frag
                #pragma unroll
                for (int t = 0; t < 4; ++t) {
                    const int nt = ntg * 4 + t;
                    const int wp = nt * 16 + n + kw; // (w + kw - 1) + 1
                    union { v4u u[2]; v16bf v; } bu;
                    const v4u* pb = (const v4u*)&xs[kh][wp][cib];
                    bu.u[0] = pb[0];
                    bu.u[1] = pb[1];
                    acc[t] = __builtin_amdgcn_wmma_f32_16x16x32_bf16(
                        false, au.v, false, bu.v, (short)0, acc[t], false, false);
                }
            }
        }
    }

    /* epilogue: bias + relu; C layout: lane n = N col, vgpr v -> M = v + 8*hh */
    const float* bp = biasws + b * COUT + mt * 16 + hh * 8;
    #pragma unroll
    for (int t = 0; t < 4; ++t) {
        const int w = (ntg * 4 + t) * 16 + n;
        #pragma unroll
        for (int v = 0; v < 8; ++v) {
            const int co = mt * 16 + hh * 8 + v;
            float val = acc[t][v] + bp[v];
            val = val > 0.f ? val : 0.f;
            out[(((long)(b * COUT + co) * HH + h) * WW) + w] = val;
        }
    }
}

extern "C" void kernel_launch(void* const* d_in, const int* in_sizes, int n_in,
                              void* d_out, int out_size, void* d_ws, size_t ws_size,
                              hipStream_t stream) {
    const float* x  = (const float*)d_in[0];
    const float* z  = (const float*)d_in[1];
    const float* W0 = (const float*)d_in[2];
    const float* b0 = (const float*)d_in[3];
    const float* W1 = (const float*)d_in[4];
    const float* b1 = (const float*)d_in[5];
    const float* W2 = (const float*)d_in[6];
    const float* b2 = (const float*)d_in[7];
    float* out = (float*)d_out;

    char*   ws     = (char*)d_ws;
    bf16_t* wsw    = (bf16_t*)ws;
    float*  biasws = (float*)(ws + WS_BIAS_OFF);
    float*  h1ws   = (float*)(ws + WS_H1_OFF);
    bf16_t* xtws   = (bf16_t*)(ws + WS_XT_OFF);

    x_to_bf16_t<<<B_ * HH, 256, 0, stream>>>(x, xtws);
    hyper_mlp<<<1, 512, 0, stream>>>(z, W0, b0, W1, b1, h1ws);
    hyper_head<<<(B_ * ROWS) / 256, 256, 0, stream>>>(W2, b2, h1ws, wsw, biasws);
    adaptive_conv_wmma<<<B_ * HH, 256, 0, stream>>>(xtws, wsw, biasws, out);
}
